// MultiHeadAttentionFromZenn_65257733095716
// MI455X (gfx1250) — matrix operand order repaired and checked by reading, not compiled, not run on hardware
//
#include <hip/hip_runtime.h>
#include <hip/hip_fp16.h>

typedef __attribute__((ext_vector_type(16))) _Float16 v16h;
typedef __attribute__((ext_vector_type(8)))  _Float16 v8h;
typedef __attribute__((ext_vector_type(4)))  _Float16 v4h;
typedef __attribute__((ext_vector_type(8)))  float    v8f;
typedef __attribute__((ext_vector_type(4)))  float    v4f;
typedef __attribute__((ext_vector_type(4)))  unsigned int uint32x4;
typedef __attribute__((ext_vector_type(8)))  int      int32x8;
typedef __attribute__((ext_vector_type(4)))  int      int32x4;

#define WMMA_F16(a,b,c) __builtin_amdgcn_wmma_f32_16x16x32_f16(false,(a),false,(b),(short)0,(c),false,false)

__device__ __forceinline__ v16h frag_cat(v8h lo, v8h hi){
  v16h r;
#pragma unroll
  for(int i=0;i<8;++i){ r[i]=lo[i]; r[i+8]=hi[i]; }
  return r;
}

// A fragment (16x32 f16) from row-major [M,K], leading dim ld (elements).
// ISA 7.12.2: lane row = lane%16; halves 0-7 -> K = k0+(lane/16)*8+i ; halves 8-15 -> +16.
__device__ __forceinline__ v16h load_frag_a(const _Float16* __restrict__ base, int ld, int k0, int lane){
  const int row = lane & 15;
  const int kb  = k0 + ((lane >> 4) << 3);
  const _Float16* p = base + (size_t)row * ld + kb;
  v8h lo = *(const v8h*)(p);
  v8h hi = *(const v8h*)(p + 16);
  return frag_cat(lo, hi);
}

// B fragment (32x16 f16) from [N,K] row-major, ld in elements.
// Lane col = lane%16; halves 0-7 -> K = k0+(lane/16)*8+i ; halves 8-15 -> +16.
__device__ __forceinline__ v16h load_frag_b(const _Float16* __restrict__ base, int ld, int k0, int lane){
  const int n  = lane & 15;
  const int kb = k0 + ((lane >> 4) << 3);
  const _Float16* p = base + (size_t)n * ld + kb;
  v8h lo = *(const v8h*)(p);
  v8h hi = *(const v8h*)(p + 16);
  return frag_cat(lo, hi);
}

// ---- Tensor Data Mover: 2-D tile load global -> LDS (D# per ISA cdna5 §8.3/8.4) ----
__device__ __forceinline__ unsigned int lds_off(const void* p){
  // low 32 bits of a generic pointer into LDS = byte offset within the WG's LDS
  return (unsigned int)(unsigned long long)p;
}

__device__ __forceinline__ void tdm_load_2d(unsigned int lds_byte, const void* gaddr,
                                            unsigned int tensor_d0, unsigned int tensor_d1,
                                            unsigned int tile_d0,   unsigned int tile_d1,
                                            unsigned int stride0_elems){
  unsigned long long ga = (unsigned long long)gaddr;
  uint32x4 g0;
  g0[0] = 1u;                                            // count=1 (valid user D#)
  g0[1] = lds_byte;                                      // lds_addr [63:32]
  g0[2] = (unsigned int)ga;                              // global_addr[31:0]
  g0[3] = (unsigned int)((ga >> 32) & 0x01FFFFFFu) | (2u << 30); // addr[56:32] | type=2
  int32x8 g1;
  g1[0] = (int)(1u << 16);                               // data_size=1 (2 bytes/elem)
  g1[1] = (int)((tensor_d0 & 0xFFFFu) << 16);            // tensor_dim0[15:0] at bits[63:48]
  g1[2] = (int)(((tensor_d0 >> 16) & 0xFFFFu) | ((tensor_d1 & 0xFFFFu) << 16));
  g1[3] = (int)(((tensor_d1 >> 16) & 0xFFFFu) | ((tile_d0 & 0xFFFFu) << 16));
  g1[4] = (int)(tile_d1 & 0xFFFFu);                      // tile_dim1 (tile_dim2 = 0)
  g1[5] = (int)stride0_elems;                            // tensor_dim0_stride[31:0]
  g1[6] = 0;                                             // stride0 hi | stride1 lo
  g1[7] = 0;                                             // stride1 hi
  int32x4 z4 = {0, 0, 0, 0};
  int32x8 z8 = {0, 0, 0, 0, 0, 0, 0, 0};
  __builtin_amdgcn_tensor_load_to_lds(g0, g1, z4, z4, z8, 0);  // 6-arg variant (clang-23)
}

// ---- x fp32 -> f16 ------------------------------------------------------
__global__ void __launch_bounds__(256) k_cvt(const float* __restrict__ in, _Float16* __restrict__ out){
  size_t i = ((size_t)blockIdx.x * 256 + threadIdx.x) * 4;
  v4f v = *(const v4f*)(in + i);
  v4h h;
#pragma unroll
  for(int j=0;j<4;++j) h[j] = (_Float16)v[j];
  *(v4h*)(out + i) = h;
}

// ---- pack Wq/Wk/Wv [H,D,DK] fp32 -> wT [3*1024][1024] f16, wT[n][d]=W[n>>7][d][n&127]
__global__ void __launch_bounds__(256) k_pack_w(const float* __restrict__ Wq, const float* __restrict__ Wk,
                                                const float* __restrict__ Wv, _Float16* __restrict__ wT){
  size_t idx = (size_t)blockIdx.x * 256 + threadIdx.x;       // 0 .. 3*2^20-1
  int region = (int)(idx >> 20);
  int rem    = (int)(idx & ((1u << 20) - 1));
  int n = rem >> 10, d = rem & 1023;
  const float* W = (region == 0) ? Wq : (region == 1) ? Wk : Wv;
  wT[idx] = (_Float16)W[(((size_t)(n >> 7) * 1024) + d) * 128 + (n & 127)];
}

// ---- fused QKV GEMM: X[8192,1024] @ wT^T -> Q,K (row-major per head-batch), V transposed
__global__ void __launch_bounds__(256) k_qkv_gemm(const _Float16* __restrict__ X, const _Float16* __restrict__ wT,
                                                  _Float16* __restrict__ qb, _Float16* __restrict__ kbuf,
                                                  _Float16* __restrict__ vT){
  const int lane = threadIdx.x & 31;
  const int w    = threadIdx.x >> 5;
  const int m0   = blockIdx.x * 128 + (w >> 1) * 32;
  const int n0   = blockIdx.y * 64  + (w & 1) * 32;
  v8f c[2][2] = {};
  const _Float16* A0 = X  + (size_t)m0 * 1024;
  const _Float16* A1 = X  + (size_t)(m0 + 16) * 1024;
  const _Float16* B0 = wT + (size_t)n0 * 1024;
  const _Float16* B1 = wT + (size_t)(n0 + 16) * 1024;
  for(int k0 = 0; k0 < 1024; k0 += 32){
    if(k0 + 128 < 1024){
      __builtin_prefetch(A0 + (size_t)(lane & 15) * 1024 + k0 + 128, 0, 3);
      __builtin_prefetch(B0 + (size_t)(lane & 15) * 1024 + k0 + 128, 0, 3);
    }
    v16h a0 = load_frag_a(A0, 1024, k0, lane);
    v16h a1 = load_frag_a(A1, 1024, k0, lane);
    v16h b0 = load_frag_b(B0, 1024, k0, lane);
    v16h b1 = load_frag_b(B1, 1024, k0, lane);
    c[0][0] = WMMA_F16(a0, b0, c[0][0]);
    c[0][1] = WMMA_F16(a0, b1, c[0][1]);
    c[1][0] = WMMA_F16(a1, b0, c[1][0]);
    c[1][1] = WMMA_F16(a1, b1, c[1][1]);
  }
  const int region = n0 >> 10;              // 0=Q 1=K 2=V (tile never straddles: 1024%32==0)
  const int rb = (lane >> 4) << 3;
#pragma unroll
  for(int i=0;i<2;++i)
#pragma unroll
    for(int j=0;j<2;++j)
#pragma unroll
      for(int r=0;r<8;++r){
        int m = m0 + i*16 + r + rb;
        int n = n0 + j*16 + (lane & 15);
        int nn = n & 1023;  int h = nn >> 7;  int kk = nn & 127;
        int b = m >> 11;    int s = m & 2047;
        _Float16 val = (_Float16)c[i][j][r];
        size_t hb = (size_t)(h * 4 + b);
        if(region == 0)      qb  [((hb * 2048 + s) << 7) + kk] = val;
        else if(region == 1) kbuf[((hb * 2048 + s) << 7) + kk] = val;
        else                 vT  [((hb * 128 + kk) << 11) + s] = val;   // transposed [d,key]
      }
}

// ---- fused attention: TDM-staged K/V in LDS (double-buffered), flash-style online softmax
__global__ void __launch_bounds__(256) k_attn(const _Float16* __restrict__ qbuf, const _Float16* __restrict__ kbuf,
                                              const _Float16* __restrict__ vT,  const unsigned char* __restrict__ mask,
                                              _Float16* __restrict__ ob){
  __shared__ _Float16 bufK[2][32][128];   // [chunk keys][d]      8 KB x2
  __shared__ _Float16 bufV[2][128][32];   // [d][chunk keys]      8 KB x2
  __shared__ _Float16 plds[8][16][32];    // per-wave P bounce    8 KB
  const int lane = threadIdx.x & 31;
  const int w    = threadIdx.x >> 5;
  const int hb   = blockIdx.y;            // h*4 + b
  const int h    = hb >> 2, b = hb & 3;
  const int q0   = blockIdx.x * 128 + w * 16;
  const _Float16* Q = qbuf + ((size_t)hb * 2048 + q0) * 128;
  const _Float16* K = kbuf + (size_t)hb * 2048 * 128;
  const _Float16* V = vT   + (size_t)hb * 128 * 2048;
  const unsigned char* M = mask + (size_t)b * 2048 * 2048;
  const bool issuer = (w == 0);           // wave 0 drives the TDM

  // prologue: DMA chunk 0 (K: 32x128 tile of [2048,128]; V: 128x32 tile of [128,2048])
  if(issuer){
    tdm_load_2d(lds_off(&bufK[0][0][0]), K, 128u, 2048u, 128u, 32u, 128u);
    tdm_load_2d(lds_off(&bufV[0][0][0]), V, 2048u, 128u, 32u, 128u, 2048u);
    __builtin_amdgcn_s_wait_tensorcnt(0);
  }
  __syncthreads();

  v16h qf[4];
#pragma unroll
  for(int d=0; d<4; ++d) qf[d] = load_frag_a(Q, 128, d * 32, lane);

  v8f o[8] = {};
  float mrow[8], lrow[8];
#pragma unroll
  for(int r=0;r<8;++r){ mrow[r] = -3.0e38f; lrow[r] = 0.f; }
  const float scale = 0.08838834764831845f;   // 1/sqrt(128)
  const int rb  = (lane >> 4) << 3;
  const int col = lane & 15;

  for(int kc = 0; kc < 64; ++kc){
    const int key0 = kc * 32;
    const int cur  = kc & 1, nxt = cur ^ 1;
    if(issuer && (kc + 1 < 64)){            // overlap next chunk's DMA with this chunk's WMMAs
      tdm_load_2d(lds_off(&bufK[nxt][0][0]), K + (size_t)(key0 + 32) * 128, 128u, 2048u, 128u, 32u, 128u);
      tdm_load_2d(lds_off(&bufV[nxt][0][0]), V + (size_t)(key0 + 32),       2048u, 128u, 32u, 128u, 2048u);
    }
    const _Float16* Kc = &bufK[cur][0][0];
    const _Float16* Vc = &bufV[cur][0][0];

    v8f s0 = {}, s1 = {};
#pragma unroll
    for(int d=0; d<4; ++d){
      v16h kf0 = load_frag_b(Kc,            128, d * 32, lane);   // keys 0-15 of chunk
      v16h kf1 = load_frag_b(Kc + 16 * 128, 128, d * 32, lane);   // keys 16-31
      s0 = WMMA_F16(qf[d], kf0, s0);
      s1 = WMMA_F16(qf[d], kf1, s1);
    }
    // scale, mask(-65504), online softmax update
#pragma unroll
    for(int r=0;r<8;++r){
      int q = q0 + r + rb;
      const unsigned char* mp = M + (size_t)q * 2048 + key0;
      float v0 = s0[r] * scale; if(mp[col])      v0 = -65504.f;
      float v1 = s1[r] * scale; if(mp[col + 16]) v1 = -65504.f;
      float mx = fmaxf(v0, v1);
#pragma unroll
      for(int off=1; off<16; off<<=1) mx = fmaxf(mx, __shfl_xor(mx, off, 32));
      float mnew  = fmaxf(mrow[r], mx);
      float alpha = __expf(mrow[r] - mnew);
      float p0 = __expf(v0 - mnew);
      float p1 = __expf(v1 - mnew);
      float ps = p0 + p1;
#pragma unroll
      for(int off=1; off<16; off<<=1) ps += __shfl_xor(ps, off, 32);
      lrow[r] = lrow[r] * alpha + ps;
      mrow[r] = mnew;
#pragma unroll
      for(int dt=0; dt<8; ++dt) o[dt][r] *= alpha;
      plds[w][r + rb][col]      = (_Float16)p0;
      plds[w][r + rb][col + 16] = (_Float16)p1;
    }
    asm volatile("s_wait_dscnt 0" ::: "memory");   // C-layout -> A-layout bounce fence
    {
      const _Float16* pp = &plds[w][lane & 15][0];
      const int kb2 = (lane >> 4) << 3;
      v16h pf = frag_cat(*(const v8h*)(pp + kb2), *(const v8h*)(pp + kb2 + 16));
#pragma unroll
      for(int dt=0; dt<8; ++dt){
        v16h vf = load_frag_b(Vc + (size_t)(dt * 16) * 32, 32, 0, lane);
        o[dt] = WMMA_F16(pf, vf, o[dt]);
      }
    }
    if(issuer) __builtin_amdgcn_s_wait_tensorcnt(0);  // next chunk's DMA landed
    __syncthreads();                                  // publish LDS to all waves
  }
  // normalize rows by l and store O (f16) head-major into [B*S, 1024]
#pragma unroll
  for(int r=0;r<8;++r){
    float inv = 1.0f / lrow[r];
    int q = q0 + r + rb;
    _Float16* orow = ob + ((size_t)(b * 2048 + q) * 1024) + h * 128;
#pragma unroll
    for(int dt=0; dt<8; ++dt) orow[dt * 16 + col] = (_Float16)(o[dt][r] * inv);
  }
}

// ---- mean over S (fp32 accumulate): om[b][j] = mean_s ob[b,s,j]
__global__ void __launch_bounds__(256) k_mean(const _Float16* __restrict__ ob, float* __restrict__ om){
  int b = blockIdx.y;
  int j = blockIdx.x * 256 + threadIdx.x;
  const _Float16* p = ob + (size_t)b * 2048 * 1024 + j;
  float s = 0.f;
  for(int t=0; t<2048; ++t) s += (float)p[(size_t)t * 1024];
  om[b * 1024 + j] = s * (1.0f / 2048.0f);
}

// ---- tiny affine layers on the mean (4 rows): t = om @ Wlin^T + blin
__global__ void __launch_bounds__(256) k_lin1(const float* __restrict__ om, const float* __restrict__ Wl,
                                              const float* __restrict__ bl, float* __restrict__ t){
  int b = blockIdx.y;
  int j = blockIdx.x * 256 + threadIdx.x;
  const float* xr = om + b * 1024;
  const float* wr = Wl + (size_t)j * 1024;
  float s = bl[j];
  for(int i=0;i<1024;++i) s += xr[i] * wr[i];
  t[b * 1024 + j] = s;
}

__global__ void __launch_bounds__(128) k_lin2(const float* __restrict__ t, const float* __restrict__ Wl,
                                              const float* __restrict__ bl, float* __restrict__ out){
  int idx = blockIdx.x * 128 + threadIdx.x;   // 512 total
  int b = idx >> 7, c = idx & 127;
  const float* xr = t  + b * 1024;
  const float* wr = Wl + (size_t)c * 1024;
  float s = bl[c];
  for(int i=0;i<1024;++i) s += xr[i] * wr[i];
  out[b * 128 + c] = s;
}

extern "C" void kernel_launch(void* const* d_in, const int* in_sizes, int n_in,
                              void* d_out, int out_size, void* d_ws, size_t ws_size,
                              hipStream_t stream) {
  const float*         x     = (const float*)d_in[0];
  const unsigned char* mask  = (const unsigned char*)d_in[1];
  const float*         Wq    = (const float*)d_in[2];
  const float*         Wk    = (const float*)d_in[3];
  const float*         Wv    = (const float*)d_in[4];
  const float*         Wlin  = (const float*)d_in[5];
  const float*         blin  = (const float*)d_in[6];
  const float*         Wlast = (const float*)d_in[7];
  const float*         blast = (const float*)d_in[8];
  float* out = (float*)d_out;

  char* ws = (char*)d_ws;
  _Float16* xh = (_Float16*)(ws + 0);          // 8192*1024 f16       = 16 MiB
  _Float16* wT = (_Float16*)(ws + 16777216);   // 3072*1024 f16       = 6 MiB
  _Float16* qb = (_Float16*)(ws + 23068672);   // 32*2048*128 f16     = 16 MiB
  _Float16* kb = (_Float16*)(ws + 39845888);   // 32*2048*128 f16     = 16 MiB
  _Float16* vT = (_Float16*)(ws + 56623104);   // 32*128*2048 f16     = 16 MiB
  _Float16* ob = (_Float16*)(ws + 73400320);   // 8192*1024 f16       = 16 MiB
  float*    om = (float*)   (ws + 90177536);   // 4*1024 f32
  float*    tt = (float*)   (ws + 90193920);   // 4*1024 f32

  k_cvt     <<<8192,          256, 0, stream>>>(x, xh);
  k_pack_w  <<<12288,         256, 0, stream>>>(Wq, Wk, Wv, wT);
  k_qkv_gemm<<<dim3(64, 48),  256, 0, stream>>>(xh, wT, qb, kb, vT);
  k_attn    <<<dim3(16, 32),  256, 0, stream>>>(qb, kb, vT, mask, ob);
  k_mean    <<<dim3(4, 4),    256, 0, stream>>>(ob, om);
  k_lin1    <<<dim3(4, 4),    256, 0, stream>>>(om, Wlin, blin, tt);
  k_lin2    <<<4,             128, 0, stream>>>(tt, Wlast, blast, out);
}
